// MIMOSSMLayer_35467839930783
// MI455X (gfx1250) — compile-verified, matrix operand order checked by reading, not run on hardware
//
#include <hip/hip_runtime.h>
#include <hip/hip_bf16.h>

// ---------------- problem constants ----------------
#define B_N   4
#define L_N   65536
#define IN_N  64
#define ST_N  64
#define OUT_N 64
#define LC    256            // chunk length (scan tile along L)
#define NC    (L_N / LC)     // 256 chunks
#define NSUB  (LC / 16)      // 16 row-subtiles of 16 per chunk

// gfx1250 async global->LDS path (ASYNCcnt-tracked), with safe fallback
#if defined(__has_builtin)
#if __has_builtin(__builtin_amdgcn_global_load_async_to_lds_b128) && \
    __has_builtin(__builtin_amdgcn_s_wait_asynccnt)
#define USE_ASYNC_LDS 1
#endif
#endif
#ifndef USE_ASYNC_LDS
#define USE_ASYNC_LDS 0
#endif

typedef __attribute__((ext_vector_type(16))) __bf16 v16bf;
typedef __attribute__((ext_vector_type(8)))  float  v8f;
typedef __attribute__((ext_vector_type(4)))  int    v4i;

__device__ inline unsigned short f2bf(float f) {
  unsigned int u = __float_as_uint(f);
  unsigned int r = u + 0x7FFFu + ((u >> 16) & 1u);   // round-to-nearest-even
  return (unsigned short)(r >> 16);
}

__device__ inline v8f wmma_bf16(v16bf a, v16bf b, v8f c) {
  return __builtin_amdgcn_wmma_f32_16x16x32_bf16(
      false, a, false, b, (short)0, c, false, false);
}

#if USE_ASYNC_LDS
typedef __attribute__((address_space(1))) v4i v4i_gbl;
typedef __attribute__((address_space(3))) v4i v4i_lds;
// per-lane 16B global -> LDS async copy
__device__ inline void async_cp16(const float* g, const float* l) {
  v4i_gbl* gp = (v4i_gbl*)(unsigned long long)(const void*)g;
  // generic LDS address: low 32 bits are the LDS byte offset (ISA aperture rule)
  v4i_lds* lp = (v4i_lds*)(unsigned int)(unsigned long long)(const void*)l;
  __builtin_amdgcn_global_load_async_to_lds_b128(gp, lp, 0, 0);
}
// stage one 16x64 fp32 subtile (4KB): 4 issues x 64 lanes x 16B
__device__ inline void issue_async_tile(const float* gsrc, const float* dst,
                                        int tid) {
#pragma unroll
  for (int q = 0; q < 4; ++q)
    async_cp16(gsrc + (q * 64 + tid) * 4, dst + (q * 64 + tid) * 4);
}
#endif

// B-fragment (K x 16 slice), source packed as [n][k] row-major (k contiguous),
// per CDNA5 16-bit operand layout: VGPR v holds K pair
//   k = k0 + ((v&4)?16:0) + (v&3)*2 + (lane>=16)*8
__device__ inline v16bf load_bfrag(const unsigned short* __restrict__ base,
                                   int n0, int k0, int lane) {
  int n = n0 + (lane & 15);
  int h = lane >> 4;
  union { unsigned int u[8]; v16bf v; } f;
#pragma unroll
  for (int r = 0; r < 8; ++r) {
    int k = k0 + ((r & 4) ? 16 : 0) + (r & 3) * 2 + h * 8;
    f.u[r] = *(const unsigned int*)(base + n * 64 + k);
  }
  return f.v;
}

// A-fragment (16 x K slice) from LDS bf16 tile [16][64]
__device__ inline v16bf load_afrag(const unsigned short* s, int k0, int lane) {
  int m = lane & 15;
  int h = lane >> 4;
  union { unsigned int u[8]; v16bf v; } f;
#pragma unroll
  for (int r = 0; r < 8; ++r) {
    int k = k0 + ((r & 4) ? 16 : 0) + (r & 3) * 2 + h * 8;
    f.u[r] = *(const unsigned int*)(s + m * 64 + k);
  }
  return f.v;
}

// ---------------- setup: ZOH discretization + bf16 packing ----------------
__global__ __launch_bounds__(64) void ssm_setup(
    const float* __restrict__ lrl, const float* __restrict__ lim,
    const float* __restrict__ Bre, const float* __restrict__ Bim,
    const float* __restrict__ Cre, const float* __restrict__ Cim,
    const float* __restrict__ ldt,
    float* __restrict__ Are, float* __restrict__ Aim,
    unsigned short* __restrict__ Bbre, unsigned short* __restrict__ Bbim,
    unsigned short* __restrict__ Cre16, unsigned short* __restrict__ Cim16) {
  int p = threadIdx.x;                    // state index / output row
  float lr = -__expf(lrl[p]);
  float li = lim[p];
  float dt = __expf(ldt[p]);
  float e  = __expf(lr * dt);
  float abr = e * __cosf(li * dt);
  float abi = e * __sinf(li * dt);
  Are[p] = abr;
  Aim[p] = abi;
  // f = (Abar - 1) / Lambda  (complex)
  float nr = abr - 1.f, ni = abi;
  float den = lr * lr + li * li;
  float fr = (nr * lr + ni * li) / den;
  float fi = (ni * lr - nr * li) / den;
#pragma unroll 4
  for (int i = 0; i < 64; ++i) {
    float br = Bre[p * 64 + i], bi = Bim[p * 64 + i];
    Bbre[p * 64 + i] = f2bf(fr * br - fi * bi);
    Bbim[p * 64 + i] = f2bf(fr * bi + fi * br);
    Cre16[p * 64 + i] = f2bf(Cre[p * 64 + i]);   // row o=p: layout [n][k]
    Cim16[p * 64 + i] = f2bf(Cim[p * 64 + i]);
  }
}

// ---------------- pass 1 / pass 3 fused chunk kernel ----------------
template <bool FINAL>
__global__ __launch_bounds__(64) void ssm_pass(
    const float* __restrict__ x, const float* __restrict__ bias_u,
    const float* __restrict__ bias_y,
    const float* __restrict__ Are, const float* __restrict__ Aim,
    const unsigned short* __restrict__ Bbre, const unsigned short* __restrict__ Bbim,
    const unsigned short* __restrict__ Cre, const unsigned short* __restrict__ Cim,
    const float* __restrict__ carryRe, const float* __restrict__ carryIm,
    float* __restrict__ outRe, float* __restrict__ outIm,
    float* __restrict__ y) {
#if USE_ASYNC_LDS
  __shared__ __align__(16) float sX[2][16 * 64];   // raw fp32 double buffer
#endif
  __shared__ __align__(16) unsigned short sU[16 * 64];  // bf16 input subtile
  __shared__ float sBu[2][16 * 64];              // Bu re/im (f32)
  __shared__ unsigned short sS[2][16 * 64];      // scan states bf16 (FINAL)
  __shared__ float sYim[16 * 64];                // imag GEMM2 partial (FINAL)
  __shared__ float sBiasU[64];
  __shared__ float sBiasY[64];

  const int tid  = threadIdx.x;
  const int wave = tid >> 5;
  const int lane = tid & 31;
  const int bid  = blockIdx.x;
  const int b    = bid / NC;
  const int nc   = bid % NC;
  const int p    = tid;                          // scan state index

  sBiasU[tid] = bias_u[tid];
  if (FINAL) sBiasY[tid] = bias_y[tid];

  const float ar = Are[p];
  const float ai = Aim[p];
  float s_re = 0.f, s_im = 0.f;
  if (FINAL) {
    size_t co = ((size_t)b * NC + nc) * 64 + p;
    s_re = carryRe[co];
    s_im = carryIm[co];
  }

  // resident B operands: wave0 = real chain, wave1 = imaginary chain
  const unsigned short* bm1 = wave ? Bbim : Bbre;
  v16bf f1[4][2];
#pragma unroll
  for (int nt = 0; nt < 4; ++nt)
#pragma unroll
    for (int t = 0; t < 2; ++t)
      f1[nt][t] = load_bfrag(bm1, nt * 16, t * 32, lane);

  v16bf f2[4][2];
  if (FINAL) {
    const unsigned short* bm2 = wave ? Cim : Cre;
#pragma unroll
    for (int nt = 0; nt < 4; ++nt)
#pragma unroll
      for (int t = 0; t < 2; ++t)
        f2[nt][t] = load_bfrag(bm2, nt * 16, t * 32, lane);
  }

  const float* xchunk = x + ((size_t)b * L_N + (size_t)nc * LC) * 64;
#if USE_ASYNC_LDS
  issue_async_tile(xchunk, sX[0], tid);          // prologue: stage subtile 0
#endif
  __syncthreads();

  for (int sub = 0; sub < NSUB; ++sub) {
    const int cur = sub & 1;
    (void)cur;
#if USE_ASYNC_LDS
    // stage NEXT subtile while this one is consumed; wait for CURRENT only
    if (sub + 1 < NSUB) {
      issue_async_tile(xchunk + (size_t)(sub + 1) * 16 * 64, sX[cur ^ 1], tid);
      __builtin_amdgcn_s_wait_asynccnt(4);
    } else {
      __builtin_amdgcn_s_wait_asynccnt(0);
    }
    __syncthreads();                             // async LDS data visible to WG
    // convert staged fp32 -> (x + bias_u) bf16 tile
    {
      unsigned int* sUw = (unsigned int*)sU;
#pragma unroll
      for (int j = 0; j < 16; j += 2) {
        int idx = tid * 16 + j;
        int col = idx & 63;
        float a0 = sX[cur][idx]     + sBiasU[col];
        float a1 = sX[cur][idx + 1] + sBiasU[col + 1];
        sUw[idx >> 1] = (unsigned int)f2bf(a0) |
                        ((unsigned int)f2bf(a1) << 16);
      }
    }
#else
    // fallback: synchronous global staging
    {
      const float* xrow = xchunk + (size_t)sub * 16 * 64;
#pragma unroll
      for (int it = 0; it < 16; ++it) {
        int idx = tid + it * 64;
        int row = idx >> 6, col = idx & 63;
        sU[idx] = f2bf(xrow[row * 64 + col] + sBiasU[col]);
      }
    }
#endif
    __syncthreads();

    // GEMM1: Bu(16x64) = u(16x64) @ Bb^T, one wave per complex component
    {
      v16bf a0 = load_afrag(sU, 0, lane);
      v16bf a1 = load_afrag(sU, 32, lane);
      int n = lane & 15, h = lane >> 4;
#pragma unroll
      for (int nt = 0; nt < 4; ++nt) {
        v8f acc = {};
        acc = wmma_bf16(a0, f1[nt][0], acc);
        acc = wmma_bf16(a1, f1[nt][1], acc);
#pragma unroll
        for (int r = 0; r < 8; ++r)
          sBu[wave][(r + 8 * h) * 64 + nt * 16 + n] = acc[r];
      }
    }
    __syncthreads();

    // complex diagonal scan: one lane per state, 16 serial steps
#pragma unroll
    for (int l = 0; l < 16; ++l) {
      float br = sBu[0][l * 64 + p];
      float bi = sBu[1][l * 64 + p];
      float nr = ar * s_re - ai * s_im + br;
      float ni = ar * s_im + ai * s_re + bi;
      s_re = nr;
      s_im = ni;
      if (FINAL) {
        sS[0][l * 64 + p] = f2bf(s_re);
        sS[1][l * 64 + p] = f2bf(s_im);
      }
    }
    __syncthreads();

    if (FINAL) {
      // GEMM2: y = relu(2*(Sre@Cre^T - Sim@Cim^T) + bias_y) / sqrt(2)
      v16bf a0 = load_afrag(sS[wave], 0, lane);
      v16bf a1 = load_afrag(sS[wave], 32, lane);
      v8f acc2[4];
      int n = lane & 15, h = lane >> 4;
#pragma unroll
      for (int nt = 0; nt < 4; ++nt) {
        v8f acc = {};
        acc = wmma_bf16(a0, f2[nt][0], acc);
        acc = wmma_bf16(a1, f2[nt][1], acc);
        acc2[nt] = acc;
      }
      if (wave == 1) {
#pragma unroll
        for (int nt = 0; nt < 4; ++nt)
#pragma unroll
          for (int r = 0; r < 8; ++r)
            sYim[(r + 8 * h) * 64 + nt * 16 + n] = acc2[nt][r];
      }
      __syncthreads();
      if (wave == 0) {
        float* yrow = y + ((size_t)b * L_N + (size_t)nc * LC + sub * 16) * 64;
#pragma unroll
        for (int nt = 0; nt < 4; ++nt)
#pragma unroll
          for (int r = 0; r < 8; ++r) {
            int m = r + 8 * h;
            int o = nt * 16 + n;
            float v = 2.f * (acc2[nt][r] - sYim[m * 64 + o]) + sBiasY[o];
            v = fmaxf(v, 0.f) * 0.70710678118654752f;
            yrow[(size_t)m * 64 + o] = v;
          }
      }
      __syncthreads();
    }
  }

  if (!FINAL) {
    size_t co = ((size_t)b * NC + nc) * 64 + p;
    outRe[co] = s_re;
    outIm[co] = s_im;
  }
}

// ---------------- pass 2: scan across chunk carries ----------------
__global__ __launch_bounds__(256) void ssm_carry(
    const float* __restrict__ lrl, const float* __restrict__ lim,
    const float* __restrict__ ldt,
    const float* __restrict__ localRe, const float* __restrict__ localIm,
    float* __restrict__ carryRe, float* __restrict__ carryIm) {
  int t = threadIdx.x;
  int b = t >> 6, p = t & 63;
  float lr = -__expf(lrl[p]);
  float li = lim[p];
  float dt = __expf(ldt[p]);
  float e  = __expf(lr * dt * (float)LC);       // A^LC = exp(Lambda*dt*LC)
  float alr = e * __cosf(li * dt * (float)LC);
  float ali = e * __sinf(li * dt * (float)LC);
  float fr = 0.f, fi = 0.f;
  for (int j = 0; j < NC; ++j) {
    size_t o = ((size_t)b * NC + j) * 64 + p;
    carryRe[o] = fr;                             // carry entering chunk j
    carryIm[o] = fi;
    float lre = localRe[o], lie = localIm[o];
    float nfr = alr * fr - ali * fi + lre;
    float nfi = alr * fi + ali * fr + lie;
    fr = nfr;
    fi = nfi;
  }
}

extern "C" void kernel_launch(void* const* d_in, const int* in_sizes, int n_in,
                              void* d_out, int out_size, void* d_ws, size_t ws_size,
                              hipStream_t stream) {
  (void)in_sizes; (void)n_in; (void)out_size; (void)ws_size;
  const float* x    = (const float*)d_in[0];
  const float* lrl  = (const float*)d_in[1];
  const float* lim  = (const float*)d_in[2];
  const float* Bre  = (const float*)d_in[3];
  const float* Bim  = (const float*)d_in[4];
  const float* Cre  = (const float*)d_in[5];
  const float* Cim  = (const float*)d_in[6];
  const float* ldt  = (const float*)d_in[7];
  const float* bu   = (const float*)d_in[8];
  const float* by   = (const float*)d_in[9];
  float* y = (float*)d_out;

  // workspace carve-out
  char* ws = (char*)d_ws;
  float* Are = (float*)ws;                         // 64 f32
  float* Aim = Are + 64;                           // 64 f32
  unsigned short* Bbre = (unsigned short*)(Aim + 64);   // 64*64 bf16
  unsigned short* Bbim = Bbre + 64 * 64;
  unsigned short* C16r = Bbim + 64 * 64;
  unsigned short* C16i = C16r + 64 * 64;
  float* localRe = (float*)(C16i + 64 * 64);       // B*NC*64 f32 each
  float* localIm = localRe + (size_t)B_N * NC * 64;
  float* carryRe = localIm + (size_t)B_N * NC * 64;
  float* carryIm = carryRe + (size_t)B_N * NC * 64;

  ssm_setup<<<1, 64, 0, stream>>>(lrl, lim, Bre, Bim, Cre, Cim, ldt,
                                  Are, Aim, Bbre, Bbim, C16r, C16i);

  ssm_pass<false><<<B_N * NC, 64, 0, stream>>>(
      x, bu, by, Are, Aim, Bbre, Bbim, C16r, C16i,
      nullptr, nullptr, localRe, localIm, nullptr);

  ssm_carry<<<1, 256, 0, stream>>>(lrl, lim, ldt, localRe, localIm,
                                   carryRe, carryIm);

  ssm_pass<true><<<B_N * NC, 64, 0, stream>>>(
      x, bu, by, Are, Aim, Bbre, Bbim, C16r, C16i,
      carryRe, carryIm, nullptr, nullptr, y);
}
